// CirculantAttention_42752104464870
// MI455X (gfx1250) — matrix operand order repaired and checked by reading, not compiled, not run on hardware
//
#include <hip/hip_runtime.h>

// ---------------------------------------------------------------------------
// CirculantAttention on MI455X (gfx1250, wave32).
// All FFTs are expressed as 128x128 DFT matmuls -> v_wmma_f32_16x16x32_bf16.
// Data stays planar [b][c][h][w] end-to-end (no transposes needed).
// ---------------------------------------------------------------------------

#define BB 8
#define CC 128
#define HH 128
#define WW 128
#define NN (HH * WW)                 // 16384 pixels per plane
#define CN ((size_t)CC * NN)         // 2,097,152 floats: one batch of [C][N]
#define BCN ((size_t)BB * CN)        // 16,777,216 floats: [B][C][N]

typedef __attribute__((ext_vector_type(16))) __bf16        v16bf;
typedef __attribute__((ext_vector_type(8)))  float         v8f;
typedef __attribute__((ext_vector_type(8)))  unsigned int  v8u;

// round-to-nearest-even f32 -> bf16, packed pair into one dword
__device__ __forceinline__ unsigned int packbf2(float lo, float hi) {
  unsigned int a = __float_as_uint(lo), b = __float_as_uint(hi);
  a = (a + 0x7FFFu + ((a >> 16) & 1u)) >> 16;
  b = (b + 0x7FFFu + ((b >> 16) & 1u)) >> 16;
  return (b << 16) | (a & 0xFFFFu);
}

__device__ __forceinline__ void zero8(v8f a[8]) {
#pragma unroll
  for (int i = 0; i < 8; ++i) {
    v8f z = {0.f, 0.f, 0.f, 0.f, 0.f, 0.f, 0.f, 0.f};
    a[i] = z;
  }
}

// ---------------------------------------------------------------------------
// One 128x128x128 GEMM pass, accumulated into acc[8] (8 N-tiles of 16x16 f32).
// 256 threads = 8 wave32s; wave w computes rows [16w, 16w+16).
// LDS (exactly 64 KB):
//   A: [m][kp] bf16-pairs, XOR-swizzled (addr = m*64 + (kp ^ (m&63)))
//      -> conflict-free fragment reads for the ISA 16-bit A layout.
//   B: [kp][n] bf16-pairs (pair along K) matching the 16-bit B layout.
// sgnA is folded into the A bf16 conversion (used to build complex combos).
// Bg2 (optional) multiplies B elementwise while staging (fused core*t).
// ---------------------------------------------------------------------------
__device__ __forceinline__ void gemm128_acc(const float* __restrict__ Ag, int lda,
                                            const float* __restrict__ Bg, long long ldb,
                                            const float* __restrict__ Bg2,
                                            float sgnA, v8f acc[8],
                                            unsigned int* __restrict__ lds) {
  __syncthreads();  // guard LDS reuse across successive passes
  // stage A (128 rows x 64 packed-K dwords)
  for (int idx = threadIdx.x; idx < 8192; idx += 256) {
    int m = idx >> 6, kp = idx & 63;
    float a0 = Ag[(size_t)m * lda + 2 * kp    ] * sgnA;
    float a1 = Ag[(size_t)m * lda + 2 * kp + 1] * sgnA;
    lds[(m << 6) | (kp ^ (m & 63))] = packbf2(a0, a1);
  }
  // stage B (64 packed-K rows x 128 cols)
  unsigned int* __restrict__ ldsB = lds + 8192;
  for (int idx = threadIdx.x; idx < 8192; idx += 256) {
    int kp = idx >> 7, n = idx & 127;
    size_t o0 = (size_t)(2 * kp) * ldb + n, o1 = o0 + (size_t)ldb;
    float b0 = Bg[o0], b1 = Bg[o1];
    if (Bg2) { b0 *= Bg2[o0]; b1 *= Bg2[o1]; }
    ldsB[idx] = packbf2(b0, b1);
  }
  __syncthreads();

  const int lane = threadIdx.x & 31;
  const int wave = threadIdx.x >> 5;
  const int hh   = lane >> 4;          // lane half (A and B: K+8 group)
  const int mrow = (wave << 4) + (lane & 15);
#pragma unroll
  for (int kb = 0; kb < 4; ++kb) {     // 4 x K=32
    v8u a;
#pragma unroll
    for (int v = 0; v < 8; ++v) {      // A frag: kp = 8*(v/4) + 4*half + (v%4)
      int kp = (kb << 4) + ((v >> 2) << 3) + (hh << 2) + (v & 3);
      a[v] = lds[(mrow << 6) | (kp ^ (mrow & 63))];
    }
    v16bf af = __builtin_bit_cast(v16bf, a);
#pragma unroll
    for (int nt = 0; nt < 8; ++nt) {
      v8u b;
#pragma unroll
      for (int v = 0; v < 8; ++v) {    // B frag: kp = 8*(v/4) + 4*half + (v%4)
        int kp = (kb << 4) + ((v >> 2) << 3) + (hh << 2) + (v & 3);
        b[v] = ldsB[(kp << 7) + (nt << 4) + (lane & 15)];
      }
      v16bf bfv = __builtin_bit_cast(v16bf, b);
      acc[nt] = __builtin_amdgcn_wmma_f32_16x16x32_bf16(
          false, af, false, bfv, (short)0, acc[nt], false, false);
    }
  }
}

// store D tiles (f32 16x16 layout: VGPR r, lanes 0-15 -> M=r, lanes 16-31 -> M=8+r)
__device__ __forceinline__ void store_scaled(float* __restrict__ out, long long ldo,
                                             const v8f acc[8], float scale) {
  const int lane = threadIdx.x & 31, wave = threadIdx.x >> 5;
  const int m0 = (wave << 4) + ((lane >> 4) << 3);
  const int nc = lane & 15;
#pragma unroll
  for (int nt = 0; nt < 8; ++nt)
#pragma unroll
    for (int r = 0; r < 8; ++r)
      out[(size_t)(m0 + r) * ldo + (nt << 4) + nc] = acc[nt][r] * scale;
}

// ---------------------------------------------------------------------------
// DFT twiddle tables: c[j*128+k] = cos(2*pi*j*k/128), s = sin(...)
// forward kernel = c - i*s, inverse kernel = c + i*s
// ---------------------------------------------------------------------------
__global__ void __launch_bounds__(256) k_init_tab(float* __restrict__ c, float* __restrict__ s) {
  int i = blockIdx.x * 256 + threadIdx.x;
  int j = i >> 7, k = i & 127;
  int t = (j * k) & 127;  // periodic: keep the argument tiny & accurate
  float ang = 0.04908738521234052f * (float)t;  // 2*pi/128
  float sv, cv;
  sincosf(ang, &sv, &cv);
  c[i] = cv; s[i] = sv;
}

// t = silu(Wg * x + bg)  (planar GEMM, M=C, N=HW, K=C per batch)
__global__ void __launch_bounds__(256) k_gate(const float* __restrict__ x,
                                              const float* __restrict__ wg,
                                              const float* __restrict__ bg,
                                              float* __restrict__ T) {
  __shared__ unsigned int lds[16384];
  const int nb = blockIdx.x, b = blockIdx.y;
  v8f acc[8]; zero8(acc);
  gemm128_acc(wg, CC, x + (size_t)b * CN + (size_t)nb * 128, NN, nullptr, 1.0f, acc, lds);
  const int lane = threadIdx.x & 31, wave = threadIdx.x >> 5;
  const int m0 = (wave << 4) + ((lane >> 4) << 3);
  const int nc = lane & 15;
#pragma unroll
  for (int nt = 0; nt < 8; ++nt)
#pragma unroll
    for (int r = 0; r < 8; ++r) {
      int m = m0 + r;
      float v = acc[nt][r] + bg[m];
      float sg = 1.0f / (1.0f + __expf(-v));
      T[(size_t)b * CN + (size_t)m * NN + (size_t)nb * 128 + (nt << 4) + nc] = v * sg;
    }
}

// forward DFT along w, real input:  Y = X * (C - iS)  => YR = X*C, YI = -X*S
__global__ void __launch_bounds__(256) k_fwd_w(const float* __restrict__ X,
                                               const float* __restrict__ tc,
                                               const float* __restrict__ ts,
                                               float* __restrict__ YR, float* __restrict__ YI) {
  __shared__ unsigned int lds[16384];
  const size_t p = blockIdx.x;
  const float* A = X + p * NN;
  v8f aR[8], aI[8]; zero8(aR); zero8(aI);
  gemm128_acc(A, WW, tc, WW, nullptr, 1.0f, aR, lds);
  gemm128_acc(A, WW, ts, WW, nullptr, 1.0f, aI, lds);
  store_scaled(YR + p * NN, WW, aR, 1.0f);
  store_scaled(YI + p * NN, WW, aI, -1.0f);
}

// forward DFT along h, complex: Z = (C - iS)*Y => ZR = C*YR + S*YI, ZI = C*YI - S*YR
__global__ void __launch_bounds__(256) k_fwd_h(const float* __restrict__ YRb,
                                               const float* __restrict__ YIb,
                                               const float* __restrict__ tc,
                                               const float* __restrict__ ts,
                                               float* __restrict__ ZR, float* __restrict__ ZI,
                                               float scale) {
  __shared__ unsigned int lds[16384];
  const size_t p = blockIdx.x;
  const float* YR = YRb + p * NN;
  const float* YI = YIb + p * NN;
  v8f aR[8], aI[8]; zero8(aR); zero8(aI);
  gemm128_acc(tc, HH, YR, WW, nullptr,  1.0f, aR, lds);
  gemm128_acc(ts, HH, YI, WW, nullptr,  1.0f, aR, lds);
  gemm128_acc(tc, HH, YI, WW, nullptr,  1.0f, aI, lds);
  gemm128_acc(ts, HH, YR, WW, nullptr, -1.0f, aI, lds);
  store_scaled(ZR + p * NN, WW, aR, scale);
  store_scaled(ZI + p * NN, WW, aI, scale);
}

// inverse DFT along w, complex: Y = S * (C + iS) => YR = SR*C - SI*S, YI = SR*S + SI*C
__global__ void __launch_bounds__(256) k_inv_w(const float* __restrict__ SRb,
                                               const float* __restrict__ SIb,
                                               const float* __restrict__ tc,
                                               const float* __restrict__ ts,
                                               float* __restrict__ YR, float* __restrict__ YI) {
  __shared__ unsigned int lds[16384];
  const size_t p = blockIdx.x;
  const float* SR = SRb + p * NN;
  const float* SI = SIb + p * NN;
  v8f aR[8], aI[8]; zero8(aR); zero8(aI);
  gemm128_acc(SR, WW, tc, WW, nullptr,  1.0f, aR, lds);
  gemm128_acc(SI, WW, ts, WW, nullptr, -1.0f, aR, lds);
  gemm128_acc(SR, WW, ts, WW, nullptr,  1.0f, aI, lds);
  gemm128_acc(SI, WW, tc, WW, nullptr,  1.0f, aI, lds);
  store_scaled(YR + p * NN, WW, aR, 1.0f);
  store_scaled(YI + p * NN, WW, aI, 1.0f);
}

// inverse DFT along h, real part only: ZR = C*YR - S*YI (input Hermitian -> real)
__global__ void __launch_bounds__(256) k_inv_h_real(const float* __restrict__ YRb,
                                                    const float* __restrict__ YIb,
                                                    const float* __restrict__ tc,
                                                    const float* __restrict__ ts,
                                                    float* __restrict__ ZR, float scale) {
  __shared__ unsigned int lds[16384];
  const size_t p = blockIdx.x;
  v8f aR[8]; zero8(aR);
  gemm128_acc(tc, HH, YRb + p * NN, WW, nullptr,  1.0f, aR, lds);
  gemm128_acc(ts, HH, YIb + p * NN, WW, nullptr, -1.0f, aR, lds);
  store_scaled(ZR + p * NN, WW, aR, scale);
}

// QKV channel GEMM in frequency space (real weight applied to re & im parts)
__global__ void __launch_bounds__(256) k_qkv(const float* __restrict__ wqkv,
                                             const float* __restrict__ Xf,
                                             float* __restrict__ out) {
  __shared__ unsigned int lds[16384];
  const int nb = blockIdx.x, rb = blockIdx.y, z = blockIdx.z;
  const int comp = z & 1, b = z >> 1;
  const float* A  = wqkv + (size_t)rb * 128 * CC;
  const float* Bm = Xf + (size_t)comp * BCN + (size_t)b * CN + (size_t)nb * 128;
  float* O = out + (size_t)comp * 3 * BCN + (size_t)b * 3 * CN
                 + (size_t)(rb * 128) * NN + (size_t)nb * 128;
  v8f acc[8]; zero8(acc);
  gemm128_acc(A, CC, Bm, NN, nullptr, 1.0f, acc, lds);
  store_scaled(O, NN, acc, 1.0f);
}

// s = conj(q) * k  (vectorized float4)
__global__ void __launch_bounds__(256) k_cqk(const float* __restrict__ qR,
                                             const float* __restrict__ qI,
                                             float* __restrict__ sR, float* __restrict__ sI) {
  size_t e = ((size_t)blockIdx.x * 256 + threadIdx.x) * 4;
  size_t b = e / CN, rem = e - b * CN;
  size_t qi = b * 3 * CN + rem, ki = qi + CN;
  float4 qr = *(const float4*)(qR + qi), qm = *(const float4*)(qI + qi);
  float4 kr = *(const float4*)(qR + ki), km = *(const float4*)(qI + ki);
  float4 orr, oii;
  orr.x = qr.x * kr.x + qm.x * km.x;  oii.x = qr.x * km.x - qm.x * kr.x;
  orr.y = qr.y * kr.y + qm.y * km.y;  oii.y = qr.y * km.y - qm.y * kr.y;
  orr.z = qr.z * kr.z + qm.z * km.z;  oii.z = qr.z * km.z - qm.z * kr.z;
  orr.w = qr.w * kr.w + qm.w * km.w;  oii.w = qr.w * km.w - qm.w * kr.w;
  *(float4*)(sR + b * CN + rem) = orr;
  *(float4*)(sI + b * CN + rem) = oii;
}

// s2 = conj(attn_f) * v
__global__ void __launch_bounds__(256) k_cav(const float* __restrict__ aR,
                                             const float* __restrict__ aI,
                                             const float* __restrict__ qkvR,
                                             const float* __restrict__ qkvI,
                                             float* __restrict__ oR, float* __restrict__ oI) {
  size_t e = ((size_t)blockIdx.x * 256 + threadIdx.x) * 4;
  size_t b = e / CN, rem = e - b * CN;
  size_t ai = b * CN + rem;
  size_t vi = b * 3 * CN + 2 * CN + rem;
  float4 ar = *(const float4*)(aR + ai), am = *(const float4*)(aI + ai);
  float4 vr = *(const float4*)(qkvR + vi), vm = *(const float4*)(qkvI + vi);
  float4 orr, oii;
  orr.x = ar.x * vr.x + am.x * vm.x;  oii.x = ar.x * vm.x - am.x * vr.x;
  orr.y = ar.y * vr.y + am.y * vm.y;  oii.y = ar.y * vm.y - am.y * vr.y;
  orr.z = ar.z * vr.z + am.z * vm.z;  oii.z = ar.z * vm.z - am.z * vr.z;
  orr.w = ar.w * vr.w + am.w * vm.w;  oii.w = ar.w * vm.w - am.w * vr.w;
  *(float4*)(oR + ai) = orr;
  *(float4*)(oI + ai) = oii;
}

// softmax over the 16384-pixel axis of one (b,c) plane, in place
__global__ void __launch_bounds__(256) k_softmax(float* __restrict__ a) {
  __shared__ float red[256];
  const size_t p = blockIdx.x;
  float* base = a + p * NN;
  const int t = threadIdx.x;
  float v[64];
#pragma unroll
  for (int i = 0; i < 64; ++i) v[i] = base[t + i * 256];
  float mx = v[0];
#pragma unroll
  for (int i = 1; i < 64; ++i) mx = fmaxf(mx, v[i]);
  red[t] = mx; __syncthreads();
  for (int s = 128; s > 0; s >>= 1) {
    if (t < s) red[t] = fmaxf(red[t], red[t + s]);
    __syncthreads();
  }
  mx = red[0]; __syncthreads();
  float sum = 0.0f;
#pragma unroll
  for (int i = 0; i < 64; ++i) { v[i] = __expf(v[i] - mx); sum += v[i]; }
  red[t] = sum; __syncthreads();
  for (int s = 128; s > 0; s >>= 1) {
    if (t < s) red[t] += red[t + s];
    __syncthreads();
  }
  const float inv = 1.0f / red[0];
#pragma unroll
  for (int i = 0; i < 64; ++i) base[t + i * 256] = v[i] * inv;
}

// out = Wp * (core .* t) + bp   (gate multiply fused into B staging)
__global__ void __launch_bounds__(256) k_proj(const float* __restrict__ core,
                                              const float* __restrict__ T,
                                              const float* __restrict__ wp,
                                              const float* __restrict__ bp,
                                              float* __restrict__ out) {
  __shared__ unsigned int lds[16384];
  const int nb = blockIdx.x, b = blockIdx.y;
  const float* B1 = core + (size_t)b * CN + (size_t)nb * 128;
  const float* B2 = T    + (size_t)b * CN + (size_t)nb * 128;
  v8f acc[8]; zero8(acc);
  gemm128_acc(wp, CC, B1, NN, B2, 1.0f, acc, lds);
  const int lane = threadIdx.x & 31, wave = threadIdx.x >> 5;
  const int m0 = (wave << 4) + ((lane >> 4) << 3);
  const int nc = lane & 15;
#pragma unroll
  for (int nt = 0; nt < 8; ++nt)
#pragma unroll
    for (int r = 0; r < 8; ++r) {
      int m = m0 + r;
      out[(size_t)b * CN + (size_t)m * NN + (size_t)nb * 128 + (nt << 4) + nc] =
          acc[nt][r] + bp[m];
    }
}

// ---------------------------------------------------------------------------
extern "C" void kernel_launch(void* const* d_in, const int* in_sizes, int n_in,
                              void* d_out, int out_size, void* d_ws, size_t ws_size,
                              hipStream_t stream) {
  (void)in_sizes; (void)n_in; (void)out_size;
  const float* x    = (const float*)d_in[0];
  const float* wqkv = (const float*)d_in[1];
  const float* wg   = (const float*)d_in[2];
  const float* bg   = (const float*)d_in[3];
  const float* wp   = (const float*)d_in[4];
  const float* bp   = (const float*)d_in[5];
  float* out = (float*)d_out;
  float* ws  = (float*)d_ws;

  // workspace map (floats): tabs | T | P1(2*BCN) | P2(2*BCN) | P3(6*BCN)
  const size_t need = (size_t)(2 * NN) + 11 * BCN;
  if (ws_size < need * sizeof(float)) return;  // ~705 MB required

  float* tc  = ws;
  float* tsn = ws + NN;
  float* T   = ws + 2 * NN;
  float* P1  = T + BCN;
  float* P2  = P1 + 2 * BCN;
  float* P3  = P2 + 2 * BCN;
  const float inv128 = 1.0f / 128.0f;  // ortho norm: 1/sqrt(H*W)

  k_init_tab<<<64, 256, 0, stream>>>(tc, tsn);
  // gate: t = silu(x @ Wg^T + bg)
  k_gate<<<dim3(128, BB), 256, 0, stream>>>(x, wg, bg, T);
  // Xf = rfft2(x, ortho) as full complex 2D DFT, scale 1/128
  k_fwd_w<<<BB * CC, 256, 0, stream>>>(x, tc, tsn, P1, P1 + BCN);
  k_fwd_h<<<BB * CC, 256, 0, stream>>>(P1, P1 + BCN, tc, tsn, P2, P2 + BCN, inv128);
  // qkv = Wqkv applied to re & im
  k_qkv<<<dim3(128, 3, BB * 2), 256, 0, stream>>>(wqkv, P2, P3);
  // s = conj(q) * k
  k_cqk<<<(unsigned)(BCN / 4 / 256), 256, 0, stream>>>(P3, P3 + 3 * BCN, P1, P1 + BCN);
  // attn = irfft2(s, ortho), real part, scale 1/128
  k_inv_w<<<BB * CC, 256, 0, stream>>>(P1, P1 + BCN, tc, tsn, P2, P2 + BCN);
  k_inv_h_real<<<BB * CC, 256, 0, stream>>>(P2, P2 + BCN, tc, tsn, P1, inv128);
  // softmax over n per (b,c)
  k_softmax<<<BB * CC, 256, 0, stream>>>(P1);
  // attn_f = rfft2(attn), backward norm (scale 1)
  k_fwd_w<<<BB * CC, 256, 0, stream>>>(P1, tc, tsn, P2, P2 + BCN);
  k_fwd_h<<<BB * CC, 256, 0, stream>>>(P2, P2 + BCN, tc, tsn, P1, P1 + BCN, 1.0f);
  // s2 = conj(attn_f) * v
  k_cav<<<(unsigned)(BCN / 4 / 256), 256, 0, stream>>>(P1, P1 + BCN, P3, P3 + 3 * BCN,
                                                       P2, P2 + BCN);
  // core = irfft2(s2, ortho), real part, scale 1/128
  k_inv_w<<<BB * CC, 256, 0, stream>>>(P2, P2 + BCN, tc, tsn, P1, P1 + BCN);
  k_inv_h_real<<<BB * CC, 256, 0, stream>>>(P1, P1 + BCN, tc, tsn, P2, inv128);
  // out = Wp * (core .* t) + bp   (planar output == [b,c,h,w] directly)
  k_proj<<<dim3(128, BB), 256, 0, stream>>>(P2, T, wp, bp, out);
}
